// _Decoder_74921409511979
// MI455X (gfx1250) — matrix operand order repaired and checked by reading, not compiled, not run on hardware
//
#include <hip/hip_runtime.h>
#include <hip/hip_bf16.h>

#define Bv 32
#define Sv 400
#define Tv 64
#define Hv 1024
#define Ev 512
#define Vv 50000
#define VEv 50050

typedef __attribute__((ext_vector_type(16))) _Float16 v16h;
typedef __attribute__((ext_vector_type(8)))  float    v8f;

// ---------------------------------------------------------------------------
// Weight conversion: f32 -> f16 (grid-stride)
// ---------------------------------------------------------------------------
__global__ void cvt16_kernel(const float* __restrict__ src, _Float16* __restrict__ dst, long n) {
    long i = (long)blockIdx.x * blockDim.x + threadIdx.x;
    long stride = (long)gridDim.x * blockDim.x;
    for (; i < n; i += stride) dst[i] = (_Float16)src[i];
}

// transpose + convert: dst[c*rows + r] = src[r*cols + c]
__global__ void cvt16_t_kernel(const float* __restrict__ src, _Float16* __restrict__ dst,
                               int rows, int cols) {
    long n = (long)rows * cols;
    long i = (long)blockIdx.x * blockDim.x + threadIdx.x;
    long stride = (long)gridDim.x * blockDim.x;
    for (; i < n; i += stride) {
        int r = (int)(i / cols), c = (int)(i % cols);
        dst[(long)c * rows + r] = src[i];
    }
}

// ---------------------------------------------------------------------------
// WMMA GEMM: C[32, N] = A[32, K](f16) @ W[N, K]^T(f16, row stride ldw)
//            (+ bias[n] if HAS_BIAS) (+ C if ADDC) (also emit f16 C if F16OUT)
// Block = 256 threads = 8 waves. Each wave computes a 32x32 output macro-tile:
// both 16-row M halves x two adjacent 16-col N tiles -> 4 accumulators.
// B fragments reused across M halves (weights read once), A fragments reused
// across N tiles. K is compile-time (512/1024) so the k-loop unrolls.
// ---------------------------------------------------------------------------
template <int K, bool HAS_BIAS, bool ADDC, bool F16OUT>
__global__ __launch_bounds__(256)
void gemm_xwt_kernel(const _Float16* __restrict__ A,
                     const _Float16* __restrict__ W, int ldw,
                     const float* __restrict__ bias,
                     float* __restrict__ C, _Float16* __restrict__ C16, int N) {
    const int lane   = threadIdx.x & 31;
    const int wave   = threadIdx.x >> 5;
    const int ntiles = N >> 4;
    const int ntile0 = (blockIdx.x * 8 + wave) * 2;
    if (ntile0 >= ntiles) return;             // whole wave exits together (EXEC stays all-1)
    const bool has2  = (ntile0 + 1) < ntiles;
    const int l16 = lane & 15;
    const int hi  = lane >> 4;                // 0 or 1

    const int n0 = ntile0 * 16 + l16;
    const int n1 = (has2 ? ntile0 + 1 : ntile0) * 16 + l16;  // clamp: in-bounds reads

    const _Float16* a0 = A + (long)l16 * K;          // M rows 0..15 fragment
    const _Float16* a1 = A + (long)(16 + l16) * K;   // M rows 16..31 fragment
    const _Float16* w0 = W + (long)n0 * ldw;
    const _Float16* w1 = W + (long)n1 * ldw;

    v8f acc00 = {}, acc01 = {}, acc10 = {}, acc11 = {};
    #pragma unroll 4
    for (int kb = 0; kb < K; kb += 32) {
        v16h af0, af1, bf0, bf1;
        // A fragment (16x32 f16, ISA layout): VGPR i holds K pair
        //   k0 = kb + (i<4?0:16) + hi*8 + 2*(i&3)  -> two contiguous 8-half runs
        #pragma unroll
        for (int i = 0; i < 8; ++i) {
            int k0 = kb + ((i & 4) << 2) + hi * 8 + ((i & 3) << 1);
            af0[2 * i]     = a0[k0];
            af0[2 * i + 1] = a0[k0 + 1];
            af1[2 * i]     = a1[k0];
            af1[2 * i + 1] = a1[k0 + 1];
        }
        // B fragment (32x16 f16): VGPR j holds K pair k0 = kb + (j<4?0:16) + hi*8 + 2*(j&3)
        #pragma unroll
        for (int j = 0; j < 8; ++j) {
            int k0 = kb + ((j & 4) << 2) + hi * 8 + ((j & 3) << 1);
            bf0[2 * j]     = w0[k0];
            bf0[2 * j + 1] = w0[k0 + 1];
            bf1[2 * j]     = w1[k0];
            bf1[2 * j + 1] = w1[k0 + 1];
        }
        acc00 = __builtin_amdgcn_wmma_f32_16x16x32_f16(false, af0, false, bf0, (short)0, acc00, false, false);
        acc01 = __builtin_amdgcn_wmma_f32_16x16x32_f16(false, af0, false, bf1, (short)0, acc01, false, false);
        acc10 = __builtin_amdgcn_wmma_f32_16x16x32_f16(false, af1, false, bf0, (short)0, acc10, false, false);
        acc11 = __builtin_amdgcn_wmma_f32_16x16x32_f16(false, af1, false, bf1, (short)0, acc11, false, false);
    }

    const float bi0 = HAS_BIAS ? bias[n0] : 0.0f;
    const float bi1 = HAS_BIAS ? bias[n1] : 0.0f;
    #pragma unroll
    for (int r = 0; r < 8; ++r) {
        const int mm0 = r + hi * 8;           // C/D ISA layout, M half 0
        const int mm1 = 16 + r + hi * 8;      // M half 1
        long i00 = (long)mm0 * N + n0;
        long i10 = (long)mm1 * N + n0;
        float v00 = acc00[r] + bi0; if (ADDC) v00 += C[i00];
        float v10 = acc10[r] + bi0; if (ADDC) v10 += C[i10];
        C[i00] = v00; if (F16OUT) C16[i00] = (_Float16)v00;
        C[i10] = v10; if (F16OUT) C16[i10] = (_Float16)v10;
        if (has2) {
            long i01 = (long)mm0 * N + n1;
            long i11 = (long)mm1 * N + n1;
            float v01 = acc01[r] + bi1; if (ADDC) v01 += C[i01];
            float v11 = acc11[r] + bi1; if (ADDC) v11 += C[i11];
            C[i01] = v01; if (F16OUT) C16[i01] = (_Float16)v01;
            C[i11] = v11; if (F16OUT) C16[i11] = (_Float16)v11;
        }
    }
}

// ---------------------------------------------------------------------------
// Attention: raw scores (q . attention[b,s,:]), masked softmax over S, context.
// One block per batch row, 256 threads. Emits context in f32 (p_gen) + f16 (GEMMs).
// ---------------------------------------------------------------------------
__global__ __launch_bounds__(256)
void attn_kernel(const float* __restrict__ query,      // [B,H]
                 const float* __restrict__ attention,  // [B,S,H]
                 const float* __restrict__ mask,       // [B,S]
                 float* __restrict__ score,            // [B,S]
                 float* __restrict__ context,          // [B,H] f32
                 _Float16* __restrict__ context16) {   // [B,H] f16
    __shared__ float sraw[Sv];
    __shared__ float red[256];
    const int b = blockIdx.x, tid = threadIdx.x;
    const float* q   = query + (long)b * Hv;
    const float* att = attention + (long)b * Sv * Hv;

    for (int s = tid; s < Sv; s += 256) {
        const float* row = att + (long)s * Hv;
        float acc = 0.f;
        for (int h = 0; h < Hv; h += 4) {
            float4 a  = *(const float4*)(row + h);
            float4 qq = *(const float4*)(q + h);
            acc += a.x * qq.x + a.y * qq.y + a.z * qq.z + a.w * qq.w;
        }
        sraw[s] = (mask[b * Sv + s] > 0.f) ? acc : -1e18f;
    }
    __syncthreads();
    float m = -3.0e38f;
    for (int s = tid; s < Sv; s += 256) m = fmaxf(m, sraw[s]);
    red[tid] = m; __syncthreads();
    for (int o = 128; o > 0; o >>= 1) { if (tid < o) red[tid] = fmaxf(red[tid], red[tid + o]); __syncthreads(); }
    m = red[0]; __syncthreads();
    float sum = 0.f;
    for (int s = tid; s < Sv; s += 256) { float e = expf(sraw[s] - m); sraw[s] = e; sum += e; }
    red[tid] = sum; __syncthreads();
    for (int o = 128; o > 0; o >>= 1) { if (tid < o) red[tid] += red[tid + o]; __syncthreads(); }
    const float inv = 1.0f / red[0]; __syncthreads();
    for (int s = tid; s < Sv; s += 256) { sraw[s] *= inv; score[b * Sv + s] = sraw[s]; }
    __syncthreads();
    for (int h = tid; h < Hv; h += 256) {
        float acc = 0.f;
        for (int s = 0; s < Sv; ++s) acc += sraw[s] * att[(long)s * Hv + h];
        context[b * Hv + h]   = acc;
        context16[b * Hv + h] = (_Float16)acc;
    }
}

// ---------------------------------------------------------------------------
// Embedding gather for step t (straight to f16 for the GEMM)
// ---------------------------------------------------------------------------
__global__ void gather_emb_kernel(const int* __restrict__ target, const float* __restrict__ emb,
                                  _Float16* __restrict__ embx16, int t) {
    const int b = blockIdx.x;
    const int tok = target[b * Tv + t];
    for (int e = threadIdx.x; e < Ev; e += blockDim.x)
        embx16[b * Ev + e] = (_Float16)emb[(long)tok * Ev + e];
}

// ---------------------------------------------------------------------------
// LSTM pointwise (gate order i,f,g,o), updates h (f32+f16) and c in place
// ---------------------------------------------------------------------------
__device__ __forceinline__ float sigm(float x) { return 1.0f / (1.0f + expf(-x)); }

__global__ void lstm_pw_kernel(const float* __restrict__ gates,  // [B,4H]
                               float* __restrict__ h, _Float16* __restrict__ h16,
                               float* __restrict__ c) {
    int idx = blockIdx.x * blockDim.x + threadIdx.x;   // B*H
    if (idx >= Bv * Hv) return;
    int b = idx / Hv, j = idx % Hv;
    const float* g = gates + (long)b * 4 * Hv;
    float gi = g[j], gf = g[Hv + j], gg = g[2 * Hv + j], go = g[3 * Hv + j];
    float c2 = sigm(gf) * c[idx] + sigm(gi) * tanhf(gg);
    c[idx] = c2;
    float hv = sigm(go) * tanhf(c2);
    h[idx]   = hv;
    h16[idx] = (_Float16)hv;
}

// ---------------------------------------------------------------------------
// p_gen = sigmoid(context.v_c + h1.v_s + lstm_in.v_i + gen_b); one block per b
// ---------------------------------------------------------------------------
__global__ __launch_bounds__(256)
void pgen_kernel(const float* __restrict__ context, const float* __restrict__ h1,
                 const float* __restrict__ lstm_in,
                 const float* __restrict__ vc, const float* __restrict__ vs,
                 const float* __restrict__ vi, const float* __restrict__ genb,
                 float* __restrict__ pgen) {
    __shared__ float red[256];
    const int b = blockIdx.x, tid = threadIdx.x;
    float acc = 0.f;
    for (int j = tid; j < Hv; j += 256)
        acc += context[b * Hv + j] * vc[j] + h1[b * Hv + j] * vs[j] + lstm_in[b * Hv + j] * vi[j];
    red[tid] = acc; __syncthreads();
    for (int o = 128; o > 0; o >>= 1) { if (tid < o) red[tid] += red[tid + o]; __syncthreads(); }
    if (tid == 0) pgen[b] = sigm(red[0] + genb[0]);
}

// ---------------------------------------------------------------------------
// Vocab softmax over VE (logits + 50 ext slots of 1e-6), scaled by p_gen,
// written directly into out[b, t, :]. One block (1024 threads) per b.
// ---------------------------------------------------------------------------
__global__ __launch_bounds__(1024)
void vocab_softmax_kernel(const float* __restrict__ logit, const float* __restrict__ pgen,
                          float* __restrict__ out, int t) {
    __shared__ float red[1024];
    const int b = blockIdx.x, tid = threadIdx.x;
    const float* lg = logit + (long)b * Vv;
    float m = 1e-6f;                                  // ext logit value participates in max
    for (int v = tid; v < Vv; v += 1024) m = fmaxf(m, lg[v]);
    red[tid] = m; __syncthreads();
    for (int o = 512; o > 0; o >>= 1) { if (tid < o) red[tid] = fmaxf(red[tid], red[tid + o]); __syncthreads(); }
    m = red[0]; __syncthreads();
    float sum = 0.f;
    for (int v = tid; v < Vv; v += 1024) sum += expf(lg[v] - m);
    if (tid == 0) sum += (float)(VEv - Vv) * expf(1e-6f - m);
    red[tid] = sum; __syncthreads();
    for (int o = 512; o > 0; o >>= 1) { if (tid < o) red[tid] += red[tid + o]; __syncthreads(); }
    const float inv = pgen[b] / red[0]; __syncthreads();
    float* o = out + ((long)b * Tv + t) * VEv;
    for (int v = tid; v < Vv; v += 1024) o[v] = expf(lg[v] - m) * inv;
    const float extv = expf(1e-6f - m) * inv;
    for (int v = Vv + tid; v < VEv; v += 1024) o[v] = extv;
}

// dist[extend_src[b,s]] += score[b,s] * (1 - p_gen)
__global__ void scatter_kernel(const int* __restrict__ extend_src, const float* __restrict__ score,
                               const float* __restrict__ pgen, float* __restrict__ out, int t) {
    const int b = blockIdx.x, tid = threadIdx.x;
    const float w = 1.0f - pgen[b];
    float* o = out + ((long)b * Tv + t) * VEv;
    for (int s = tid; s < Sv; s += blockDim.x)
        atomicAdd(&o[extend_src[b * Sv + s]], score[b * Sv + s] * w);
}

// out = log(out + 1e-8) over [B, VE] slice t
__global__ __launch_bounds__(1024)
void log_kernel(float* __restrict__ out, int t) {
    long idx = (long)blockIdx.x * blockDim.x + threadIdx.x;
    if (idx >= (long)Bv * VEv) return;
    int b = (int)(idx / VEv), v = (int)(idx % VEv);
    long p = ((long)b * Tv + t) * VEv + v;
    out[p] = logf(out[p] + 1e-8f);
}

// ---------------------------------------------------------------------------
// Host: orchestrate the whole decoder scan on `stream`
// ---------------------------------------------------------------------------
extern "C" void kernel_launch(void* const* d_in, const int* in_sizes, int n_in,
                              void* d_out, int out_size, void* d_ws, size_t ws_size,
                              hipStream_t stream) {
    const float* attention = (const float*)d_in[0];
    const float* attn_mask = (const float*)d_in[1];
    const int*   extendsrc = (const int*)  d_in[2];
    const float* h0        = (const float*)d_in[3];
    const float* c0        = (const float*)d_in[4];
    const int*   target    = (const int*)  d_in[5];
    const float* emb       = (const float*)d_in[6];
    const float* attn_w    = (const float*)d_in[7];
    const float* W_in      = (const float*)d_in[8];
    const float* b_in      = (const float*)d_in[9];
    const float* W_o1      = (const float*)d_in[10];
    const float* b_o1      = (const float*)d_in[11];
    const float* W_o2      = (const float*)d_in[12];
    const float* b_o2      = (const float*)d_in[13];
    const float* v_c       = (const float*)d_in[14];
    const float* v_s       = (const float*)d_in[15];
    const float* v_i       = (const float*)d_in[16];
    const float* gen_b     = (const float*)d_in[17];
    const float* W_ih0     = (const float*)d_in[18];
    const float* W_hh0     = (const float*)d_in[19];
    const float* b_ih0     = (const float*)d_in[20];
    const float* b_hh0     = (const float*)d_in[21];
    const float* W_ih1     = (const float*)d_in[22];
    const float* W_hh1     = (const float*)d_in[23];
    const float* b_ih1     = (const float*)d_in[24];
    const float* b_hh1     = (const float*)d_in[25];
    float* out = (float*)d_out;

    // ---- workspace bump allocator (256B aligned) ----
    char* wsp = (char*)d_ws;
    auto alloc = [&](size_t bytes) -> void* {
        void* p = (void*)wsp;
        wsp += (bytes + 255) & ~(size_t)255;
        return p;
    };
    _Float16* attnT16 = (_Float16*)alloc((size_t)Hv * Hv * 2);            // attn_w transposed
    _Float16* win16   = (_Float16*)alloc((size_t)Hv * (Ev + Hv) * 2);
    _Float16* wo116   = (_Float16*)alloc((size_t)Ev * 2 * Hv * 2);
    _Float16* wo216   = (_Float16*)alloc((size_t)Vv * Ev * 2);
    _Float16* wih016  = (_Float16*)alloc((size_t)4 * Hv * Hv * 2);
    _Float16* whh016  = (_Float16*)alloc((size_t)4 * Hv * Hv * 2);
    _Float16* wih116  = (_Float16*)alloc((size_t)4 * Hv * Hv * 2);
    _Float16* whh116  = (_Float16*)alloc((size_t)4 * Hv * Hv * 2);
    float*    hbuf    = (float*)   alloc((size_t)2 * Bv * Hv * 4);   // [2,B,H] f32
    _Float16* h16     = (_Float16*)alloc((size_t)2 * Bv * Hv * 2);   // [2,B,H] f16
    float*    cbuf    = (float*)   alloc((size_t)2 * Bv * Hv * 4);
    float*    query   = (float*)   alloc((size_t)Bv * Hv * 4);
    float*    score   = (float*)   alloc((size_t)Bv * Sv * 4);
    float*    context = (float*)   alloc((size_t)Bv * Hv * 4);
    _Float16* ctx16   = (_Float16*)alloc((size_t)Bv * Hv * 2);
    _Float16* embx16  = (_Float16*)alloc((size_t)Bv * Ev * 2);
    float*    lstm_in = (float*)   alloc((size_t)Bv * Hv * 4);
    _Float16* lin16   = (_Float16*)alloc((size_t)Bv * Hv * 2);
    float*    gates   = (float*)   alloc((size_t)Bv * 4 * Hv * 4);
    float*    dec_out = (float*)   alloc((size_t)Bv * Ev * 4);
    _Float16* dec16   = (_Float16*)alloc((size_t)Bv * Ev * 2);
    float*    logit   = (float*)   alloc((size_t)Bv * Vv * 4);
    float*    pgen    = (float*)   alloc((size_t)Bv * 4);
    (void)ws_size; (void)in_sizes; (void)n_in; (void)out_size;

    // ---- per-call weight conversion to f16 (L2-resident working set) ----
    auto cvt = [&](const float* src, _Float16* dst, long n) {
        int grid = (int)((n + 255) / 256); if (grid > 65535) grid = 65535;
        cvt16_kernel<<<grid, 256, 0, stream>>>(src, dst, n);
    };
    {
        long n = (long)Hv * Hv;
        cvt16_t_kernel<<<(int)((n + 255) / 256), 256, 0, stream>>>(attn_w, attnT16, Hv, Hv);
    }
    cvt(W_in,  win16,  (long)Hv * (Ev + Hv));
    cvt(W_o1,  wo116,  (long)Ev * 2 * Hv);
    cvt(W_o2,  wo216,  (long)Vv * Ev);
    cvt(W_ih0, wih016, (long)4 * Hv * Hv);
    cvt(W_hh0, whh016, (long)4 * Hv * Hv);
    cvt(W_ih1, wih116, (long)4 * Hv * Hv);
    cvt(W_hh1, whh116, (long)4 * Hv * Hv);

    // ---- init state (f32 copy + f16 shadow of h) ----
    hipMemcpyAsync(hbuf, h0, (size_t)2 * Bv * Hv * 4, hipMemcpyDeviceToDevice, stream);
    hipMemcpyAsync(cbuf, c0, (size_t)2 * Bv * Hv * 4, hipMemcpyDeviceToDevice, stream);
    cvt(h0, h16, (long)2 * Bv * Hv);

    float*    h0s  = hbuf;               // layer 0 hidden [B,H] f32
    float*    h1s  = hbuf + Bv * Hv;     // layer 1 hidden (top) f32
    _Float16* h0s16 = h16;
    _Float16* h1s16 = h16 + Bv * Hv;
    float*    c0s  = cbuf;
    float*    c1s  = cbuf + Bv * Hv;

    // dispatch to compile-time K / bias / addC / f16-out variants
    auto gemm = [&](const _Float16* A, const _Float16* W, int ldw, const float* bias,
                    float* C, _Float16* C16, int K, int N, int addC) {
        dim3 grid((N / 16 + 15) / 16);
#define DG(KC, HB, AC, FO) gemm_xwt_kernel<KC, HB, AC, FO><<<grid, 256, 0, stream>>>(A, W, ldw, bias, C, C16, N)
        if (K == 512) {
            if (bias) { if (addC) { if (C16) DG(512, true,  true,  true); else DG(512, true,  true,  false); }
                        else      { if (C16) DG(512, true,  false, true); else DG(512, true,  false, false); } }
            else      { if (addC) { if (C16) DG(512, false, true,  true); else DG(512, false, true,  false); }
                        else      { if (C16) DG(512, false, false, true); else DG(512, false, false, false); } }
        } else {
            if (bias) { if (addC) { if (C16) DG(1024, true,  true,  true); else DG(1024, true,  true,  false); }
                        else      { if (C16) DG(1024, true,  false, true); else DG(1024, true,  false, false); } }
            else      { if (addC) { if (C16) DG(1024, false, true,  true); else DG(1024, false, true,  false); }
                        else      { if (C16) DG(1024, false, false, true); else DG(1024, false, false, false); } }
        }
#undef DG
    };

    for (int t = 0; t < Tv; ++t) {
        // attention uses pre-update top-layer state
        gemm(h1s16, attnT16, Hv, nullptr, query, nullptr, Hv, Hv, 0);      // query = h1 @ attn_w
        attn_kernel<<<Bv, 256, 0, stream>>>(query, attention, attn_mask, score, context, ctx16);

        // lstm_in = [emb[tok], context] @ W_in^T + b_in   (split-K, no concat)
        gather_emb_kernel<<<Bv, 256, 0, stream>>>(target, emb, embx16, t);
        gemm(embx16, win16,      Ev + Hv, b_in,    lstm_in, nullptr, Ev, Hv, 0);
        gemm(ctx16,  win16 + Ev, Ev + Hv, nullptr, lstm_in, lin16,   Hv, Hv, 1);

        // LSTM layer 0
        gemm(lin16, wih016, Hv, b_ih0, gates, nullptr, Hv, 4 * Hv, 0);
        gemm(h0s16, whh016, Hv, b_hh0, gates, nullptr, Hv, 4 * Hv, 1);
        lstm_pw_kernel<<<(Bv * Hv + 255) / 256, 256, 0, stream>>>(gates, h0s, h0s16, c0s);
        // LSTM layer 1 (x = new h0, recurrent = old h1)
        gemm(h0s16, wih116, Hv, b_ih1, gates, nullptr, Hv, 4 * Hv, 0);
        gemm(h1s16, whh116, Hv, b_hh1, gates, nullptr, Hv, 4 * Hv, 1);
        lstm_pw_kernel<<<(Bv * Hv + 255) / 256, 256, 0, stream>>>(gates, h1s, h1s16, c1s);

        // dec_out = [h1, context] @ W_o1^T + b_o1   (split-K; f16 copy feeds logits)
        gemm(h1s16, wo116,      2 * Hv, b_o1,    dec_out, nullptr, Hv, Ev, 0);
        gemm(ctx16, wo116 + Hv, 2 * Hv, nullptr, dec_out, dec16,   Hv, Ev, 1);

        // logits = dec_out @ W_o2^T + b_o2  (dominant GEMM, N=50000=3125*16)
        gemm(dec16, wo216, Ev, b_o2, logit, nullptr, Ev, Vv, 0);

        // p_gen, vocab softmax (scaled by p_gen, straight into out), copy scatter, log
        pgen_kernel<<<Bv, 256, 0, stream>>>(context, h1s, lstm_in, v_c, v_s, v_i, gen_b, pgen);
        vocab_softmax_kernel<<<Bv, 1024, 0, stream>>>(logit, pgen, out, t);
        scatter_kernel<<<Bv, 256, 0, stream>>>(extendsrc, score, pgen, out, t);
        log_kernel<<<(int)(((long)Bv * VEv + 1023) / 1024), 1024, 0, stream>>>(out, t);
    }
}
